// Attention_72258529788331
// MI455X (gfx1250) — matrix operand order repaired and checked by reading, not compiled, not run on hardware
//
#include <hip/hip_runtime.h>
#include <hip/hip_bf16.h>
#include <math.h>

typedef __attribute__((ext_vector_type(16))) _Float16 v16h;
typedef __attribute__((ext_vector_type(8)))  _Float16 v8h;
typedef __attribute__((ext_vector_type(4)))  _Float16 v4h;
typedef __attribute__((ext_vector_type(8)))  float    v8f;
typedef __attribute__((ext_vector_type(4)))  int      v4i;

__device__ __forceinline__ v16h cat8(v8h lo, v8h hi) {
    return __builtin_shufflevector(lo, hi, 0,1,2,3,4,5,6,7,8,9,10,11,12,13,14,15);
}

#define WMMA_F16(A,B,C) __builtin_amdgcn_wmma_f32_16x16x32_f16(false,(A),false,(B),(short)0,(C),false,false)

// ---- CDNA5 async global->LDS copy (ASYNCcnt path), with safe fallback -----
#if defined(__has_builtin)
#  if __has_builtin(__builtin_amdgcn_global_load_async_to_lds_b128)
#    define HAVE_ASYNC_LDS 1
#  endif
#endif
#ifndef HAVE_ASYNC_LDS
#  define HAVE_ASYNC_LDS 0
#endif

typedef __attribute__((address_space(1))) v4i g_v4i;
typedef __attribute__((address_space(3))) v4i l_v4i;

__device__ __forceinline__ void async_copy16B(const _Float16* g, _Float16* l) {
#if HAVE_ASYNC_LDS
    __builtin_amdgcn_global_load_async_to_lds_b128((g_v4i*)g, (l_v4i*)l, 0, 0);
#else
    *(v8h*)l = *(const v8h*)g;
#endif
}

__device__ __forceinline__ void async_wait_all() {
#if HAVE_ASYNC_LDS
#  if __has_builtin(__builtin_amdgcn_s_wait_asynccnt)
    __builtin_amdgcn_s_wait_asynccnt(0);
#  else
    asm volatile("s_wait_asynccnt 0x0" ::: "memory");
#  endif
#endif
}

// ---------------------------------------------------------------- constants
#define BATCH 2
#define SEQ   2048
#define CDIM  1024
#define HEADS 16
#define HDIM  64
#define MROWS (BATCH*SEQ)   // 4096

// ---------------------------------------------------------------- fp32 -> fp16
__global__ __launch_bounds__(256) void cvt_f32_f16(const float* __restrict__ x,
                                                   _Float16* __restrict__ y, int n) {
    int i = (blockIdx.x * blockDim.x + threadIdx.x) * 4;
    if (i < n) {
        float4 v = *(const float4*)(x + i);
        v4h o; o.x = (_Float16)v.x; o.y = (_Float16)v.y; o.z = (_Float16)v.z; o.w = (_Float16)v.w;
        *(v4h*)(y + i) = o;
    }
}

// fp32 RxC -> fp16 transposed CxR   (y[c][r] = (half)x[r][c])
__global__ __launch_bounds__(256) void cvt_transpose_f16(const float* __restrict__ x,
                                                         _Float16* __restrict__ y,
                                                         int R, int Ccols) {
    __shared__ float tile[32][33];
    const int c0 = blockIdx.x * 32, r0 = blockIdx.y * 32;
    const int tx = threadIdx.x & 31, ty = threadIdx.x >> 5;   // 32 x 8
    #pragma unroll
    for (int i = 0; i < 32; i += 8)
        tile[ty + i][tx] = x[(size_t)(r0 + ty + i) * Ccols + c0 + tx];
    __syncthreads();
    #pragma unroll
    for (int i = 0; i < 32; i += 8)
        y[(size_t)(c0 + ty + i) * R + r0 + tx] = (_Float16)tile[tx][ty + i];
}

// ---------------------------------------------------------------- WMMA GEMM
// C[M,N] = A[M,K] * Bt^T, where Bt is the PRE-TRANSPOSED weight [N,K] f16.
// mode 0: write f32 row-major [M,N] (final projection).
// mode 1: f16 head-split  dst[((b*H+h)*T+t)*64+d] = acc*scale   (Q, K)
// mode 2: f16 d-major     dst[((b*H+h)*64+d)*T+t] = acc         (V transposed)
#define TM 128
#define TN 128
#define TK 32
__global__ __launch_bounds__(256) void gemm_f16(const _Float16* __restrict__ A,
                                                const _Float16* __restrict__ Bt,
                                                void* __restrict__ Out,
                                                int M, int N, int K,
                                                int mode, float scale) {
    __shared__ _Float16 As[TM][TK + 8];   // 128 x 40
    __shared__ _Float16 Bs[TN][TK + 8];   // 128 x 40  ([n][k])

    const int tid  = threadIdx.x;
    const int lane = tid & 31;
    const int w    = tid >> 5;            // 8 waves
    const int wm   = w & 3;               // 4 waves along M (32 rows each)
    const int wn   = w >> 2;              // 2 waves along N (64 cols each)
    const int m0   = blockIdx.y * TM;
    const int n0   = blockIdx.x * TN;
    const int lhalf = lane >> 4;
    const int l16   = lane & 15;
    const int kbA   = lhalf * 8;          // A-frag K base (interleaved 8+8)
    const int kbB   = lhalf * 8;          // B-frag K base (interleaved 8+8)

    v8f acc[2][4];
    #pragma unroll
    for (int i = 0; i < 2; i++)
        #pragma unroll
        for (int j = 0; j < 4; j++) acc[i][j] = (v8f)0.f;

    const int srow = tid >> 1;            // 0..127
    const int sc0  = (tid & 1) * 16;      // 0 / 16

    for (int k0 = 0; k0 < K; k0 += TK) {
        // stage A tile + B tile: all contiguous 16B async copies
        {
            const _Float16* ga = A  + (size_t)(m0 + srow) * K + k0 + sc0;
            async_copy16B(ga,     &As[srow][sc0]);
            async_copy16B(ga + 8, &As[srow][sc0 + 8]);
            const _Float16* gb = Bt + (size_t)(n0 + srow) * K + k0 + sc0;
            async_copy16B(gb,     &Bs[srow][sc0]);
            async_copy16B(gb + 8, &Bs[srow][sc0 + 8]);
        }
        async_wait_all();
        __syncthreads();

        v16h afrag[2], bfrag[4];
        #pragma unroll
        for (int i = 0; i < 2; i++) {
            int row = wm * 32 + i * 16 + l16;
            afrag[i] = cat8(*(const v8h*)&As[row][kbA], *(const v8h*)&As[row][kbA + 16]);
        }
        #pragma unroll
        for (int j = 0; j < 4; j++) {
            int col = wn * 64 + j * 16 + l16;
            bfrag[j] = cat8(*(const v8h*)&Bs[col][kbB], *(const v8h*)&Bs[col][kbB + 16]);
        }
        #pragma unroll
        for (int i = 0; i < 2; i++)
            #pragma unroll
            for (int j = 0; j < 4; j++)
                acc[i][j] = WMMA_F16(afrag[i], bfrag[j], acc[i][j]);
        __syncthreads();
    }

    // epilogue: C/D layout -> lane l16 = col, rows lhalf*8 + r
    #pragma unroll
    for (int i = 0; i < 2; i++)
        #pragma unroll
        for (int j = 0; j < 4; j++) {
            int n = n0 + wn * 64 + j * 16 + l16;
            if (mode == 0) {
                float* O = (float*)Out;
                #pragma unroll
                for (int r = 0; r < 8; r++) {
                    int m = m0 + wm * 32 + i * 16 + lhalf * 8 + r;
                    O[(size_t)m * N + n] = acc[i][j][r];
                }
            } else if (mode == 1) {
                _Float16* O = (_Float16*)Out;
                int h = n >> 6, d = n & 63;
                #pragma unroll
                for (int r = 0; r < 8; r++) {
                    int m = m0 + wm * 32 + i * 16 + lhalf * 8 + r;
                    int b = m >> 11, t = m & (SEQ - 1);
                    O[(((size_t)b * HEADS + h) * SEQ + t) * HDIM + d] =
                        (_Float16)(acc[i][j][r] * scale);
                }
            } else {
                _Float16* O = (_Float16*)Out;
                int h = n >> 6, d = n & 63;
                #pragma unroll
                for (int r = 0; r < 8; r++) {
                    int m = m0 + wm * 32 + i * 16 + lhalf * 8 + r;
                    int b = m >> 11, t = m & (SEQ - 1);
                    O[(((size_t)b * HEADS + h) * HDIM + d) * SEQ + t] =
                        (_Float16)acc[i][j][r];
                }
            }
        }
}

// ---------------------------------------------------------------- flash attention with sink
// Q,K: f16 [B,H,T,64] (Q pre-scaled 1/8).  VT: f16 [B,H,64,T] (d-major).
// Out: f16 [B,T,C] at cols h*64+d.  Block = 128 thr = 4 waves, 64 q rows.
__global__ __launch_bounds__(128) void flash_attn(const _Float16* __restrict__ Q,
                                                  const _Float16* __restrict__ Kh,
                                                  const _Float16* __restrict__ VT,
                                                  const float* __restrict__ sink,
                                                  _Float16* __restrict__ Ao) {
    __shared__ _Float16 Ks[64][72];        // [key][d]
    __shared__ _Float16 Vt[64][72];        // [d][key]
    __shared__ _Float16 Ps[4][16][72];     // wave-private P tiles [q][key]

    const int tid  = threadIdx.x;
    const int lane = tid & 31;
    const int w    = tid >> 5;
    const int qb   = blockIdx.x * 64;
    const int h    = blockIdx.y;
    const int b    = blockIdx.z;
    const int lhalf = lane >> 4;
    const int l16   = lane & 15;
    const int kbA   = lhalf * 8;
    const int kbB   = lhalf * 8;
    const size_t head  = ((size_t)b * HEADS + h) * SEQ;          // rows of Q/K
    const size_t vbase = ((size_t)b * HEADS + h) * HDIM * SEQ;   // VT base

    const int qt = qb + w * 16;
    const _Float16* qp = Q + (head + qt + l16) * HDIM;
    v16h qf0 = cat8(*(const v8h*)(qp + kbA),      *(const v8h*)(qp + kbA + 16));
    v16h qf1 = cat8(*(const v8h*)(qp + 32 + kbA), *(const v8h*)(qp + 32 + kbA + 16));

    const float s0 = sink[h];
    float m_i[8], l_i[8];
    v8f o[4];
    #pragma unroll
    for (int r = 0; r < 8; r++) { m_i[r] = s0; l_i[r] = 1.0f; }
    #pragma unroll
    for (int j = 0; j < 4; j++) o[j] = (v8f)0.f;

    const int srow = tid >> 1;            // 0..63
    const int sc0  = (tid & 1) * 32;      // 0 / 32

    for (int k0 = 0; k0 < qb + 64; k0 += 64) {
        // stage K tile [key][d] and V tile [d][key]: contiguous async copies
        {
            const _Float16* gk = Kh + (head + k0 + srow) * HDIM + sc0;
            const _Float16* gv = VT + vbase + (size_t)srow * SEQ + k0 + sc0;
            #pragma unroll
            for (int i = 0; i < 4; i++) {
                async_copy16B(gk + i * 8, &Ks[srow][sc0 + i * 8]);
                async_copy16B(gv + i * 8, &Vt[srow][sc0 + i * 8]);
            }
        }
        async_wait_all();
        __syncthreads();

        // scores S = Q*K^T for 4 key sub-tiles (64 keys)
        float p[4][8];
        float newm[8];
        #pragma unroll
        for (int r = 0; r < 8; r++) newm[r] = m_i[r];
        #pragma unroll
        for (int kt = 0; kt < 4; kt++) {
            v8f s = (v8f)0.f;
            const int krow = kt * 16 + l16;
            v16h kf0 = cat8(*(const v8h*)&Ks[krow][kbB],      *(const v8h*)&Ks[krow][kbB + 16]);
            v16h kf1 = cat8(*(const v8h*)&Ks[krow][32 + kbB], *(const v8h*)&Ks[krow][32 + kbB + 16]);
            s = WMMA_F16(qf0, kf0, s);
            s = WMMA_F16(qf1, kf1, s);
            int key = k0 + kt * 16 + l16;
            #pragma unroll
            for (int r = 0; r < 8; r++) {
                int q = qt + lhalf * 8 + r;
                float sv = (key <= q) ? s[r] : -1e30f;   // causal mask
                p[kt][r] = sv;
                newm[r] = fmaxf(newm[r], sv);
            }
        }
        // row max across the 16-lane half
        #pragma unroll
        for (int off = 1; off < 16; off <<= 1)
            #pragma unroll
            for (int r = 0; r < 8; r++)
                newm[r] = fmaxf(newm[r], __shfl_xor(newm[r], off, 32));

        float rs[8];
        #pragma unroll
        for (int r = 0; r < 8; r++) {
            float sc = __expf(m_i[r] - newm[r]);
            l_i[r] *= sc;
            m_i[r]  = newm[r];
            #pragma unroll
            for (int j = 0; j < 4; j++) o[j][r] *= sc;
            rs[r] = 0.0f;
        }
        #pragma unroll
        for (int kt = 0; kt < 4; kt++)
            #pragma unroll
            for (int r = 0; r < 8; r++) {
                float e = __expf(p[kt][r] - m_i[r]);
                rs[r] += e;
                Ps[w][lhalf * 8 + r][kt * 16 + l16] = (_Float16)e;
            }
        #pragma unroll
        for (int off = 1; off < 16; off <<= 1)
            #pragma unroll
            for (int r = 0; r < 8; r++)
                rs[r] += __shfl_xor(rs[r], off, 32);
        #pragma unroll
        for (int r = 0; r < 8; r++) l_i[r] += rs[r];

        // O += P * V  (P reloaded from wave-private LDS in A-layout)
        v16h pa0 = cat8(*(const v8h*)&Ps[w][l16][kbA],      *(const v8h*)&Ps[w][l16][kbA + 16]);
        v16h pa1 = cat8(*(const v8h*)&Ps[w][l16][32 + kbA], *(const v8h*)&Ps[w][l16][32 + kbA + 16]);
        #pragma unroll
        for (int j = 0; j < 4; j++) {
            const int drow = j * 16 + l16;
            v16h vf0 = cat8(*(const v8h*)&Vt[drow][kbB],      *(const v8h*)&Vt[drow][kbB + 16]);
            v16h vf1 = cat8(*(const v8h*)&Vt[drow][32 + kbB], *(const v8h*)&Vt[drow][32 + kbB + 16]);
            o[j] = WMMA_F16(pa0, vf0, o[j]);
            o[j] = WMMA_F16(pa1, vf1, o[j]);
        }
        __syncthreads();
    }

    #pragma unroll
    for (int j = 0; j < 4; j++) {
        int n = h * HDIM + j * 16 + l16;
        #pragma unroll
        for (int r = 0; r < 8; r++) {
            int q = qt + lhalf * 8 + r;
            float inv = 1.0f / l_i[r];
            Ao[((size_t)b * SEQ + q) * CDIM + n] = (_Float16)(o[j][r] * inv);
        }
    }
}

// ---------------------------------------------------------------- launcher
extern "C" void kernel_launch(void* const* d_in, const int* in_sizes, int n_in,
                              void* d_out, int out_size, void* d_ws, size_t ws_size,
                              hipStream_t stream) {
    const float* x    = (const float*)d_in[0];
    const float* W_Q  = (const float*)d_in[1];
    const float* W_K  = (const float*)d_in[2];
    const float* W_V  = (const float*)d_in[3];
    const float* W_o  = (const float*)d_in[4];
    const float* sink = (const float*)d_in[5];
    float* out = (float*)d_out;

    const int NX = MROWS * CDIM;     // 4,194,304
    const int NW = CDIM * CDIM;      // 1,048,576

    _Float16* ws  = (_Float16*)d_ws;
    size_t off = 0;
    _Float16* xh  = ws + off; off += NX;
    _Float16* Wqt = ws + off; off += NW;   // pre-transposed weights [N][K]
    _Float16* Wkt = ws + off; off += NW;
    _Float16* Wvt = ws + off; off += NW;
    _Float16* Wot = ws + off; off += NW;
    _Float16* Qh  = ws + off; off += NX;   // [B,H,T,64]
    _Float16* Kh  = ws + off; off += NX;   // [B,H,T,64]
    _Float16* Vt  = ws + off; off += NX;   // [B,H,64,T]
    _Float16* Ah  = ws + off; off += NX;   // [B,T,C]

    cvt_f32_f16<<<NX / 1024, 256, 0, stream>>>(x, xh, NX);
    dim3 gt(CDIM / 32, CDIM / 32);
    cvt_transpose_f16<<<gt, 256, 0, stream>>>(W_Q, Wqt, CDIM, CDIM);
    cvt_transpose_f16<<<gt, 256, 0, stream>>>(W_K, Wkt, CDIM, CDIM);
    cvt_transpose_f16<<<gt, 256, 0, stream>>>(W_V, Wvt, CDIM, CDIM);
    cvt_transpose_f16<<<gt, 256, 0, stream>>>(W_o, Wot, CDIM, CDIM);

    dim3 gg(CDIM / TN, MROWS / TM);  // (8, 32)
    gemm_f16<<<gg, 256, 0, stream>>>(xh, Wqt, Qh, MROWS, CDIM, CDIM, 1, 0.125f);
    gemm_f16<<<gg, 256, 0, stream>>>(xh, Wkt, Kh, MROWS, CDIM, CDIM, 1, 1.0f);
    gemm_f16<<<gg, 256, 0, stream>>>(xh, Wvt, Vt, MROWS, CDIM, CDIM, 2, 1.0f);

    dim3 ga(SEQ / 64, HEADS, BATCH); // (32, 16, 2)
    flash_attn<<<ga, 128, 0, stream>>>(Qh, Kh, Vt, sink, Ah);

    gemm_f16<<<gg, 256, 0, stream>>>(Ah, Wot, out, MROWS, CDIM, CDIM, 0, 1.0f);
}
